// minerva_wrapper2_60000693125634
// MI455X (gfx1250) — matrix operand order repaired and checked by reading, not compiled, not run on hardware
//
#include <hip/hip_runtime.h>

typedef __attribute__((ext_vector_type(16))) _Float16 v16h;
typedef __attribute__((ext_vector_type(8)))  float    v8f;

#define F_DIM 768
#define E_DIM 512
#define B_DIM 4096
#define N_DIM 16384

// sim kernel tiling: block = 8 waves; each wave owns a 16-row M-tile (128 rows/block),
// all waves share a 128-column Dh panel staged in LDS (double buffered).
#define SIM_ROWS 128
#define SIM_COLS 128
#define LDS_STRIDE 80                 // bytes per 32-half row: 64B data + 16B pad (bank spread)
#define LDS_BUF (SIM_COLS * LDS_STRIDE)  // 10240 B per buffer

union H8 { uint4 u4; _Float16 h[8]; };

// ---------------- CDNA5 async copy (global -> LDS), ASYNCcnt tracked ----------------
__device__ inline void async_copy_b128(const void* gptr, unsigned lds_off) {
    asm volatile("global_load_async_to_lds_b128 %0, %1, off"
                 :
                 : "v"(lds_off), "v"(gptr)
                 : "memory");
}
__device__ inline void wait_async0() {
    asm volatile("s_wait_asynccnt 0" ::: "memory");
}

// ---- fragment loaders (16x16x32 f16 WMMA operands, contiguous k order, same for A and B) ----
// A (16x32, MxK): lane m = lane&15, g = lane>>4.
//   elems i : K = k0 + g*16 + i
__device__ inline v16h load_a_frag_f16(const _Float16* __restrict__ rowPtr, int k0, int g) {
    H8 lo, hi;
    lo.u4 = *(const uint4*)(rowPtr + k0 + g * 16);
    hi.u4 = *(const uint4*)(rowPtr + k0 + g * 16 + 8);
    v16h a;
#pragma unroll
    for (int i = 0; i < 8; ++i) { a[i] = lo.h[i]; a[i + 8] = hi.h[i]; }
    return a;
}

__device__ inline v16h load_a_frag_f32(const float* __restrict__ rowPtr, int k0, int g) {
    const float4* p0 = (const float4*)(rowPtr + k0 + g * 16);
    const float4* p1 = (const float4*)(rowPtr + k0 + g * 16 + 8);
    float4 x0 = p0[0], x1 = p0[1], y0 = p1[0], y1 = p1[1];
    v16h a;
    a[0]  = (_Float16)x0.x; a[1]  = (_Float16)x0.y; a[2]  = (_Float16)x0.z; a[3]  = (_Float16)x0.w;
    a[4]  = (_Float16)x1.x; a[5]  = (_Float16)x1.y; a[6]  = (_Float16)x1.z; a[7]  = (_Float16)x1.w;
    a[8]  = (_Float16)y0.x; a[9]  = (_Float16)y0.y; a[10] = (_Float16)y0.z; a[11] = (_Float16)y0.w;
    a[12] = (_Float16)y1.x; a[13] = (_Float16)y1.y; a[14] = (_Float16)y1.z; a[15] = (_Float16)y1.w;
    return a;
}

// B (32x16, KxN): lane n = lane&15, g = lane>>4. elems i: K = k0 + g*16 + i.
__device__ inline v16h load_b_frag_f32(const float* __restrict__ rowPtr, int k0, int g) {
    const float4* p = (const float4*)(rowPtr + k0 + g * 16);
    float4 q0 = p[0], q1 = p[1], q2 = p[2], q3 = p[3];
    v16h b;
    b[0]  = (_Float16)q0.x; b[1]  = (_Float16)q0.y; b[2]  = (_Float16)q0.z; b[3]  = (_Float16)q0.w;
    b[4]  = (_Float16)q1.x; b[5]  = (_Float16)q1.y; b[6]  = (_Float16)q1.z; b[7]  = (_Float16)q1.w;
    b[8]  = (_Float16)q2.x; b[9]  = (_Float16)q2.y; b[10] = (_Float16)q2.z; b[11] = (_Float16)q2.w;
    b[12] = (_Float16)q3.x; b[13] = (_Float16)q3.y; b[14] = (_Float16)q3.z; b[15] = (_Float16)q3.w;
    return b;
}

// ---------- Phase 1/2: E = l2norm(A @ Gw^T + Gb), fp16 out ----------
__global__ void __launch_bounds__(256)
embed_norm_kernel(const float* __restrict__ A, const float* __restrict__ Gw,
                  const float* __restrict__ Gb, _Float16* __restrict__ Out)
{
    __shared__ float s_sq[16];
    const int tid   = threadIdx.x;
    const int wave  = tid >> 5;
    const int lane  = tid & 31;
    const int lhalf = lane & 15;
    const int g     = lane >> 4;
    const int m0    = blockIdx.x * 16;
    const int e_base = wave * 64;

    if (tid < 16) s_sq[tid] = 0.0f;
    __syncthreads();

    v8f vzero = {0.f, 0.f, 0.f, 0.f, 0.f, 0.f, 0.f, 0.f};
    v8f acc[4];
#pragma unroll
    for (int t = 0; t < 4; ++t) acc[t] = vzero;

    const float* arow = A + (size_t)(m0 + lhalf) * F_DIM;

    for (int k0 = 0; k0 < F_DIM; k0 += 32) {
        v16h afrag = load_a_frag_f32(arow, k0, g);
#pragma unroll
        for (int t = 0; t < 4; ++t) {
            const float* wrow = Gw + (size_t)(e_base + t * 16 + lhalf) * F_DIM;
            v16h bfrag = load_b_frag_f32(wrow, k0, g);
            acc[t] = __builtin_amdgcn_wmma_f32_16x16x32_f16(
                false, afrag, false, bfrag, (short)0, acc[t], false, false);
        }
    }

#pragma unroll
    for (int t = 0; t < 4; ++t) {
        float bias = Gb[e_base + t * 16 + lhalf];
#pragma unroll
        for (int i = 0; i < 8; ++i) acc[t][i] += bias;
    }

    float sq[8];
#pragma unroll
    for (int i = 0; i < 8; ++i) {
        float s = 0.f;
#pragma unroll
        for (int t = 0; t < 4; ++t) { float v = acc[t][i]; s += v * v; }
        sq[i] = s;
    }
#pragma unroll
    for (int m = 1; m <= 8; m <<= 1) {
#pragma unroll
        for (int i = 0; i < 8; ++i) sq[i] += __shfl_xor(sq[i], m, 32);
    }
    if (lhalf == 0) {
#pragma unroll
        for (int i = 0; i < 8; ++i) atomicAdd(&s_sq[i + g * 8], sq[i]);
    }
    __syncthreads();

    float scale[8];
#pragma unroll
    for (int i = 0; i < 8; ++i) {
        float nrm = sqrtf(s_sq[i + g * 8]);
        scale[i] = 1.0f / fmaxf(nrm, 1e-12f);
    }

#pragma unroll
    for (int t = 0; t < 4; ++t) {
        int e = e_base + t * 16 + lhalf;
#pragma unroll
        for (int i = 0; i < 8; ++i) {
            int row = i + g * 8;
            Out[(size_t)(m0 + row) * E_DIM + e] = (_Float16)(acc[t][i] * scale[i]);
        }
    }
}

// ---------- Phase 3: echo[b] += sum_n (Xn[b].Dn[n])^3 * (2 r[n]-1) ----------
// LDS-staged (async copy, double buffered) Dh panel shared by all 8 waves.
__global__ void __launch_bounds__(256)
sim_echo_kernel(const _Float16* __restrict__ Xh, const _Float16* __restrict__ Dh,
                const float* __restrict__ r, float* __restrict__ echo)
{
    __shared__ __align__(16) char smem[2 * LDS_BUF];

    const int tid   = threadIdx.x;
    const int wave  = tid >> 5;
    const int lane  = tid & 31;
    const int lhalf = lane & 15;
    const int g     = lane >> 4;
    const int m0    = blockIdx.x * SIM_ROWS + wave * 16;   // wave-private 16 rows
    const int nbase = blockIdx.y * SIM_COLS;

    // staging assignment: thread t stages 32B of Dh row (nbase + t/2), half (t&1)
    const int srow  = tid >> 1;
    const int spart = tid & 1;
    const _Float16* gstage = Dh + (size_t)(nbase + srow) * E_DIM + spart * 16;
    const unsigned lds0 = (unsigned)(size_t)(void*)smem;
    const unsigned lrow = lds0 + (unsigned)(srow * LDS_STRIDE + spart * 32);

    // prologue: stage k0 = 0 into buffer 0
    async_copy_b128(gstage + 0, lrow + 0);
    async_copy_b128(gstage + 8, lrow + 16);

    v8f vzero = {0.f, 0.f, 0.f, 0.f, 0.f, 0.f, 0.f, 0.f};
    v8f acc[8];
#pragma unroll
    for (int t = 0; t < 8; ++t) acc[t] = vzero;

    const _Float16* xrow = Xh + (size_t)(m0 + lhalf) * E_DIM;

    for (int k = 0; k < E_DIM / 32; ++k) {
        const int k0  = k * 32;
        const int buf = k & 1;

        wait_async0();       // our staged panel for `buf` has landed
        __syncthreads();     // everyone staged + everyone done reading buf^1

        if (k < E_DIM / 32 - 1) {  // stage next panel into buf^1
            const _Float16* gs = gstage + (k0 + 32);
            const unsigned  lo = lrow + (unsigned)((buf ^ 1) * LDS_BUF);
            async_copy_b128(gs + 0, lo + 0);
            async_copy_b128(gs + 8, lo + 16);
        }

        v16h afrag = load_a_frag_f16(xrow, k0, g);
        const char* bbase = smem + buf * LDS_BUF;
#pragma unroll
        for (int t = 0; t < 8; ++t) {
            const char* lp = bbase + (t * 16 + lhalf) * LDS_STRIDE + g * 32;
            H8 lo8, hi8;
            lo8.u4 = *(const uint4*)lp;
            hi8.u4 = *(const uint4*)(lp + 16);
            v16h bfrag;
#pragma unroll
            for (int i = 0; i < 8; ++i) { bfrag[i] = lo8.h[i]; bfrag[i + 8] = hi8.h[i]; }
            acc[t] = __builtin_amdgcn_wmma_f32_16x16x32_f16(
                false, afrag, false, bfrag, (short)0, acc[t], false, false);
        }
    }

    // fused epilogue: a^3 * r2, reduce over columns (wave-private rows)
    float p[8] = {0.f, 0.f, 0.f, 0.f, 0.f, 0.f, 0.f, 0.f};
#pragma unroll
    for (int t = 0; t < 8; ++t) {
        float rw = 2.0f * r[nbase + t * 16 + lhalf] - 1.0f;
#pragma unroll
        for (int i = 0; i < 8; ++i) {
            float a = acc[t][i];
            p[i] += a * a * a * rw;
        }
    }
#pragma unroll
    for (int m = 1; m <= 8; m <<= 1) {
#pragma unroll
        for (int i = 0; i < 8; ++i) p[i] += __shfl_xor(p[i], m, 32);
    }
    if (lhalf == 0) {
#pragma unroll
        for (int i = 0; i < 8; ++i) atomicAdd(&echo[m0 + i + g * 8], p[i]);
    }
}

__global__ void zero_kernel(float* __restrict__ p, int n) {
    int i = blockIdx.x * blockDim.x + threadIdx.x;
    if (i < n) p[i] = 0.0f;
}

__global__ void finalize_kernel(const float* __restrict__ echo,
                                const float* __restrict__ hw, const float* __restrict__ hb,
                                float* __restrict__ out, int n) {
    int i = blockIdx.x * blockDim.x + threadIdx.x;
    if (i < n) {
        float l = echo[i] * hw[0] + hb[0];
        out[i]     = l;                        // logits
        out[n + i] = 1.0f / (1.0f + expf(-l)); // preds
    }
}

extern "C" void kernel_launch(void* const* d_in, const int* in_sizes, int n_in,
                              void* d_out, int out_size, void* d_ws, size_t ws_size,
                              hipStream_t stream) {
    const float* X  = (const float*)d_in[0];
    const float* D  = (const float*)d_in[1];
    const float* r  = (const float*)d_in[2];
    const float* Gw = (const float*)d_in[3];
    const float* Gb = (const float*)d_in[4];
    const float* hw = (const float*)d_in[5];
    const float* hb = (const float*)d_in[6];
    float* out = (float*)d_out;

    char* ws = (char*)d_ws;
    _Float16* Xh = (_Float16*)ws;                                    // 4 MB
    _Float16* Dh = (_Float16*)(ws + (size_t)B_DIM * E_DIM * 2);      // 16 MB
    float*  echo = (float*)(ws + (size_t)B_DIM * E_DIM * 2
                               + (size_t)N_DIM * E_DIM * 2);         // 16 KB

    embed_norm_kernel<<<B_DIM / 16, 256, 0, stream>>>(X, Gw, Gb, Xh);
    embed_norm_kernel<<<N_DIM / 16, 256, 0, stream>>>(D, Gw, Gb, Dh);
    zero_kernel<<<(B_DIM + 255) / 256, 256, 0, stream>>>(echo, B_DIM);
    sim_echo_kernel<<<dim3(B_DIM / SIM_ROWS, N_DIM / SIM_COLS), 256, 0, stream>>>(Xh, Dh, r, echo);
    finalize_kernel<<<(B_DIM + 255) / 256, 256, 0, stream>>>(echo, hw, hb, out, B_DIM);
}
